// Layer_6871947673892
// MI455X (gfx1250) — compile-verified, matrix operand order checked by reading, not run on hardware
//
#include <hip/hip_runtime.h>

// ---------------------------------------------------------------------------
// LSTM forward for MI455X (gfx1250, wave32, WMMA bf16 16x16x32, f32 accum)
//
// Phase 0: convert x, W, R, y0 to bf16 (RNE) in workspace.
// Phase 1: gates[T*B,4H] = x_bf16 @ W_bf16^T + bW + bR
//          128x128 block-tiled WMMA GEMM, LDS double-buffered k-slabs staged
//          with CDNA5 async-to-LDS loads (ASYNCcnt).
// Phase 2: ONE persistent kernel for the whole recurrence:
//          - each block owns a (16 batch x 16 hidden) tile for all T steps
//          - its R slice (4 gates x 16 rows x H) is async-staged to LDS once
//          - 4 waves split K, LDS-reduce, fused LSTM cell elementwise
//          - device-scope atomic barrier (+s_sleep spin) between timesteps
// ---------------------------------------------------------------------------

typedef __attribute__((ext_vector_type(16))) __bf16     v16bf;
typedef __attribute__((ext_vector_type(8)))  float      v8f;
typedef __attribute__((ext_vector_type(4)))  unsigned   v4u;
typedef __attribute__((__vector_size__(16))) int        v4i_vs;   // async builtin pointee
typedef unsigned short ushort_t;

#if defined(__has_builtin)
#if __has_builtin(__builtin_amdgcn_global_load_async_to_lds_b128) && \
    __has_builtin(__builtin_amdgcn_s_wait_asynccnt)
#define USE_ASYNC_LDS 1
#endif
#endif
#ifndef USE_ASYNC_LDS
#define USE_ASYNC_LDS 0
#endif

__device__ __forceinline__ void async_cp16(const ushort_t* src, ushort_t* dst_lds) {
#if USE_ASYNC_LDS
    __builtin_amdgcn_global_load_async_to_lds_b128(
        (__attribute__((address_space(1))) v4i_vs*)src,
        (__attribute__((address_space(3))) v4i_vs*)dst_lds,
        /*offset=*/0, /*cpol=*/0);
#else
    *(v4u*)dst_lds = *(const v4u*)src;
#endif
}

__device__ __forceinline__ void async_wait_all() {
#if USE_ASYNC_LDS
    __builtin_amdgcn_s_wait_asynccnt(0);
#endif
}

__device__ __forceinline__ ushort_t f32_to_bf16(float f) {
    union { float f; unsigned u; } cv; cv.f = f;
    unsigned u = cv.u;
    unsigned r = (u + 0x7FFFu + ((u >> 16) & 1u)) >> 16;  // round-to-nearest-even
    return (ushort_t)r;
}

// 16x32 bf16 fragment from a row-major matrix (ld in elements); works for
// global or LDS-backed pointers.  Lane l: row (row0 + l%16);
// elems 0..7 = K k0+8*(l/16)+0..7, elems 8..15 = K k0+16+8*(l/16)+0..7.
__device__ __forceinline__ v16bf load_frag(const ushort_t* p,
                                           int ld, int row0, int k0, int lane) {
    int m = lane & 15, half = lane >> 4;
    const ushort_t* r = p + (size_t)(row0 + m) * ld + (k0 + half * 8);
    union { v4u u[2]; v16bf v; } un;
    un.u[0] = *(const v4u*)(r);
    un.u[1] = *(const v4u*)(r + 16);
    return un.v;
}

__device__ __forceinline__ v8f wmma_bf16(v16bf a, v16bf b, v8f c) {
    return __builtin_amdgcn_wmma_f32_16x16x32_bf16(
        false, a, false, b, (short)0, c, false, false);
}

__device__ __forceinline__ float sigmoidf_fast(float x) {
    return 1.0f / (1.0f + __expf(-x));
}

// ---------------------------------------------------------------------------
// Phase 0: f32 -> bf16 conversion (grid-stride)
// ---------------------------------------------------------------------------
__global__ void cvt_bf16_kernel(const float* __restrict__ in,
                                ushort_t* __restrict__ out, size_t n) {
    size_t i      = (size_t)blockIdx.x * blockDim.x + threadIdx.x;
    size_t stride = (size_t)gridDim.x * blockDim.x;
    for (; i < n; i += stride) out[i] = f32_to_bf16(in[i]);
}

// ---------------------------------------------------------------------------
// Phase 1: gates = x @ W^T + (bW + bR)
// ---------------------------------------------------------------------------
__global__ __launch_bounds__(256) void gates_gemm_kernel(
    const ushort_t* __restrict__ x_bf,
    const ushort_t* __restrict__ w_bf,
    const float* __restrict__ bW, const float* __restrict__ bR,
    float* __restrict__ gates, int K, int N) {

    __shared__ __align__(16) ushort_t As[2][128][32];   // 2 x 8 KB
    __shared__ __align__(16) ushort_t Bs[2][128][32];   // 2 x 8 KB

    const int tid  = threadIdx.x;
    const int lane = tid & 31;
    const int wave = tid >> 5;
    const int wm   = wave >> 2;            // 0..1
    const int wn   = wave & 3;             // 0..3

    const int nb_count  = N >> 7;
    const int mb        = (int)blockIdx.x / nb_count;
    const int nb        = (int)blockIdx.x - mb * nb_count;
    const int row_block = mb << 7;
    const int col_block = nb << 7;

    v8f acc[4][2];
#pragma unroll
    for (int i = 0; i < 4; ++i)
#pragma unroll
        for (int j = 0; j < 2; ++j) acc[i][j] = (v8f){};

    // 128x32 slab stager: 512 16B chunks, 256 threads x 2
    auto stage = [&](const ushort_t* gsrc, int rb, int k0, ushort_t* lds) {
#pragma unroll
        for (int c = 0; c < 2; ++c) {
            int chunk = c * 256 + tid;
            int row = chunk >> 2;
            int kc  = (chunk & 3) * 8;
            async_cp16(gsrc + (size_t)(rb + row) * K + k0 + kc,
                       lds + row * 32 + kc);
        }
    };

    const int nslab = K >> 5;
    int buf = 0;
    stage(x_bf, row_block, 0, &As[0][0][0]);
    stage(w_bf, col_block, 0, &Bs[0][0][0]);
    async_wait_all();
    __syncthreads();

    for (int ks = 0; ks < nslab; ++ks) {
        const int nbuf = buf ^ 1;
        if (ks + 1 < nslab) {               // prefetch next slab while computing
            stage(x_bf, row_block, (ks + 1) << 5, &As[nbuf][0][0]);
            stage(w_bf, col_block, (ks + 1) << 5, &Bs[nbuf][0][0]);
        }
        v16bf af[4], bfv[2];
#pragma unroll
        for (int i = 0; i < 4; ++i)
            af[i] = load_frag(&As[buf][0][0], 32, wm * 64 + i * 16, 0, lane);
#pragma unroll
        for (int j = 0; j < 2; ++j)
            bfv[j] = load_frag(&Bs[buf][0][0], 32, wn * 32 + j * 16, 0, lane);
#pragma unroll
        for (int i = 0; i < 4; ++i)
#pragma unroll
            for (int j = 0; j < 2; ++j)
                acc[i][j] = wmma_bf16(af[i], bfv[j], acc[i][j]);
        async_wait_all();
        __syncthreads();
        buf = nbuf;
    }

    const int lcol = lane & 15;
    const int lrow = (lane >> 4) << 3;
#pragma unroll
    for (int j = 0; j < 2; ++j) {
        int col = col_block + wn * 32 + j * 16 + lcol;
        float bias = bW[col] + bR[col];
#pragma unroll
        for (int i = 0; i < 4; ++i) {
            int rbase = row_block + wm * 64 + i * 16 + lrow;
#pragma unroll
            for (int r = 0; r < 8; ++r)
                __builtin_nontemporal_store(
                    acc[i][j][r] + bias,
                    &gates[(size_t)(rbase + r) * N + col]);
        }
    }
}

// ---------------------------------------------------------------------------
// Phase 2: persistent recurrence kernel.
//   grid = (B/16, H/16) = (4,64) blocks, 128 threads (4 waves) each.
//   Dynamic LDS: R slice [4 gates][16 rows][H + 8 pad] bf16 (~129 KB).
// ---------------------------------------------------------------------------
#define RS_PAD 8
__global__ __launch_bounds__(128) void lstm_persistent_kernel(
    ushort_t* __restrict__ ybf0,             // [B,H] bf16 state ping
    ushort_t* __restrict__ ybf1,             // [B,H] bf16 state pong
    const float* __restrict__ c0,            // [B,H] f32
    const ushort_t* __restrict__ r_bf,       // [4H,H] bf16
    const float* __restrict__ gates,         // [T,B,4H] f32
    float* __restrict__ y_all, float* __restrict__ y_all2,
    float* __restrict__ c_all,
    float* __restrict__ yn, float* __restrict__ cn,
    unsigned* __restrict__ bar,
    int T, int B, int H) {

    extern __shared__ __align__(16) ushort_t Rs[];      // [4][16][H+RS_PAD]
    __shared__ __align__(16) float part[4][4][8][32];   // 16 KB partials

    const int tid  = threadIdx.x;
    const int lane = tid & 31;
    const int wave = tid >> 5;               // 0..3 : K split
    const int mt = blockIdx.x, nt = blockIdx.y;
    const int row0 = mt << 4, col0 = nt << 4;
    const int RS_LD = H + RS_PAD;
    const int H4 = H << 2;
    const size_t BH = (size_t)B * H;
    const unsigned nblocks = gridDim.x * gridDim.y;

    // ---- stage this block's R slice into LDS once (async) ----
    {
        const int chunks_per_row = H >> 3;               // 16B chunks
        const int chunks_per_gate = chunks_per_row << 4; // 16 rows
        const int total = chunks_per_gate << 2;          // 4 gates
        for (int c = tid; c < total; c += 128) {
            int g   = c / chunks_per_gate;
            int rem = c - g * chunks_per_gate;
            int row = rem / chunks_per_row;
            int kc  = (rem - row * chunks_per_row) * 8;
            async_cp16(r_bf + (size_t)(g * H + col0 + row) * H + kc,
                       Rs + (size_t)(g * 16 + row) * RS_LD + kc);
        }
        async_wait_all();
        __syncthreads();
    }

    const int kspan = H >> 2;                // K per wave
    const int kbeg  = wave * kspan;

    for (int t = 0; t < T; ++t) {
        const ushort_t* yprev = (t & 1) ? ybf1 : ybf0;
        ushort_t*       ynext = (t & 1) ? ybf0 : ybf1;
        const float* cprev   = (t == 0) ? c0 : (c_all + (size_t)(t - 1) * BH);
        const float* gates_t = gates + (size_t)t * B * H4;

        // ---- K-split WMMA accumulation (R from LDS, y_prev from L2) ----
        v8f ai = {}, af = {}, ag = {}, ao = {};
        for (int k = kbeg; k < kbeg + kspan; k += 32) {
            v16bf a  = load_frag(yprev, H, row0, k, lane);
            v16bf bi = load_frag(Rs,              RS_LD, 0,  k, lane);
            v16bf bf = load_frag(Rs + 16 * RS_LD, RS_LD, 0,  k, lane);
            v16bf bg = load_frag(Rs + 32 * RS_LD, RS_LD, 0,  k, lane);
            v16bf bo = load_frag(Rs + 48 * RS_LD, RS_LD, 0,  k, lane);
            ai = wmma_bf16(a, bi, ai);
            af = wmma_bf16(a, bf, af);
            ag = wmma_bf16(a, bg, ag);
            ao = wmma_bf16(a, bo, ao);
        }
#pragma unroll
        for (int r = 0; r < 8; ++r) {
            part[wave][0][r][lane] = ai[r];
            part[wave][1][r][lane] = af[r];
            part[wave][2][r][lane] = ag[r];
            part[wave][3][r][lane] = ao[r];
        }
        __syncthreads();

        // ---- reduce + fused LSTM cell: 256 elements, 128 threads x 2 ----
#pragma unroll
        for (int e2 = 0; e2 < 2; ++e2) {
            int e = tid + (e2 << 7);
            int row = e >> 4, colL = e & 15;
            int li = colL + ((row >> 3) << 4);
            int r  = row & 7;
            float gi = 0.f, gf = 0.f, gg = 0.f, go = 0.f;
#pragma unroll
            for (int w = 0; w < 4; ++w) {
                gi += part[w][0][r][li];
                gf += part[w][1][r][li];
                gg += part[w][2][r][li];
                go += part[w][3][r][li];
            }
            int rowg = row0 + row;
            int colg = col0 + colL;
            size_t gb = (size_t)rowg * H4 + colg;
            gi += __builtin_nontemporal_load(&gates_t[gb]);
            gf += __builtin_nontemporal_load(&gates_t[gb + H]);
            gg += __builtin_nontemporal_load(&gates_t[gb + 2 * (size_t)H]);
            go += __builtin_nontemporal_load(&gates_t[gb + 3 * (size_t)H]);
            float ig = sigmoidf_fast(gi);
            float fg = sigmoidf_fast(gf);
            float g  = tanhf(gg);
            float og = sigmoidf_fast(go);
            size_t idx = (size_t)rowg * H + colg;
            float c = fg * cprev[idx] + ig * g;
            float y = og * tanhf(c);
            __builtin_nontemporal_store(y, &y_all[(size_t)t * BH + idx]);
            __builtin_nontemporal_store(y, &y_all2[(size_t)t * BH + idx]);
            c_all[(size_t)t * BH + idx] = c;      // re-read next step: RT
            ynext[idx] = f32_to_bf16(y);
            if (t == T - 1) { yn[idx] = y; cn[idx] = c; }
        }

        // ---- device-wide barrier between timesteps ----
        __threadfence();
        __syncthreads();
        if (tid == 0) {
            __hip_atomic_fetch_add(bar, 1u, __ATOMIC_ACQ_REL,
                                   __HIP_MEMORY_SCOPE_AGENT);
            unsigned target = nblocks * (unsigned)(t + 1);
            while (__hip_atomic_load(bar, __ATOMIC_ACQUIRE,
                                     __HIP_MEMORY_SCOPE_AGENT) < target)
                __builtin_amdgcn_s_sleep(2);
        }
        __syncthreads();
        __threadfence();
    }
}

// ---------------------------------------------------------------------------
extern "C" void kernel_launch(void* const* d_in, const int* in_sizes, int n_in,
                              void* d_out, int out_size, void* d_ws, size_t ws_size,
                              hipStream_t stream) {
    const int T = 512, B = 64, I = 1024, H = 1024;
    const int fourH = 4 * H;
    const size_t M  = (size_t)T * B;
    const size_t BH = (size_t)B * H;

    const float* x  = (const float*)d_in[0];
    const float* y0 = (const float*)d_in[1];
    const float* c0 = (const float*)d_in[2];
    const float* W  = (const float*)d_in[3];
    const float* R  = (const float*)d_in[4];
    const float* bW = (const float*)d_in[5];
    const float* bR = (const float*)d_in[6];

    float* out    = (float*)d_out;
    float* y_all  = out;                       // [T,B,H]
    float* yn     = y_all + (size_t)T * BH;
    float* cn     = yn + BH;
    float* y_all2 = cn + BH;                   // [T,B,H]
    float* c_all  = y_all2 + (size_t)T * BH;   // [T,B,H]

    char* ws = (char*)d_ws;
    unsigned* bar  = (unsigned*)ws;  ws += 256;                    // barrier ctr
    ushort_t* x_bf = (ushort_t*)ws;  ws += M * I * sizeof(ushort_t);
    ushort_t* w_bf = (ushort_t*)ws;  ws += (size_t)fourH * I * sizeof(ushort_t);
    ushort_t* r_bf = (ushort_t*)ws;  ws += (size_t)fourH * H * sizeof(ushort_t);
    ushort_t* ybf0 = (ushort_t*)ws;  ws += BH * sizeof(ushort_t);
    ushort_t* ybf1 = (ushort_t*)ws;  ws += BH * sizeof(ushort_t);
    float*    gates = (float*)ws;    // [T*B, 4H] f32

    hipMemsetAsync(bar, 0, sizeof(unsigned), stream);

    cvt_bf16_kernel<<<4096, 256, 0, stream>>>(x,  x_bf, M * I);
    cvt_bf16_kernel<<<1024, 256, 0, stream>>>(W,  w_bf, (size_t)fourH * I);
    cvt_bf16_kernel<<<1024, 256, 0, stream>>>(R,  r_bf, (size_t)fourH * H);
    cvt_bf16_kernel<<<64,   256, 0, stream>>>(y0, ybf0, BH);

    // Phase 1: (M/128) x (4H/128) = 256 x 32 = 8192 blocks of 256 threads
    {
        unsigned blocks = (unsigned)((M >> 7) * (fourH >> 7));
        gates_gemm_kernel<<<blocks, 256, 0, stream>>>(x_bf, w_bf, bW, bR,
                                                      gates, I, fourH);
    }

    // Phase 2: one persistent kernel for all T steps
    {
        dim3 grid(B / 16, H / 16);                       // 256 blocks
        size_t rs_bytes = (size_t)4 * 16 * (H + RS_PAD) * sizeof(ushort_t);
        lstm_persistent_kernel<<<grid, 128, rs_bytes, stream>>>(
            ybf0, ybf1, c0, r_bf, gates, y_all, y_all2, c_all,
            yn, cn, bar, T, B, H);
    }
}
